// LCAOInteraction_53326313947774
// MI455X (gfx1250) — compile-verified, matrix operand order checked by reading, not run on hardware
//
#include <hip/hip_runtime.h>
#include <hip/hip_bf16.h>

// LCAO interaction layer for MI455X (gfx1250, wave32, WMMA).
// Memory-bound workload (~20 GFLOP vs ~0.9 GB traffic). All dense GEMMs via
// v_wmma_f32_16x16x32_f16. LDS weights and inter-layer staging are stored in
// WMMA *fragment order* (2x ds_load_b128 per fragment); B fragments are
// batch-loaded into distinct registers ahead of the WMMA chain so LDS latency
// overlaps the matrix pipe instead of serializing it. Big (E,9,64)
// intermediate kept in f16 (halves gather traffic; 192MB L2 captures the
// triplet-gather reuse).

#define N_NODES 10000
#define N_EDGES 100000
#define N_TRI   200000
#define NORB    9
#define HDIM    128
#define CDIM    64

typedef __attribute__((ext_vector_type(16))) _Float16 v16h;
typedef __attribute__((ext_vector_type(2)))  _Float16 v2h;
typedef __attribute__((ext_vector_type(8)))  float    v8f;

// K index held by element t of a v16h A/B fragment for this half of the wave.
__device__ __forceinline__ int kmap(int t, int half) {
  return (t < 8) ? (half * 8 + t) : (16 + half * 8 + (t - 8));
}

// Slot (in halves) of matrix element (row m, col) inside an A-fragment-ordered
// 16 x 64 staging tile: fragment ks = col/32, lane = half(col)*16 + m.
__device__ __forceinline__ int frag_slot(int m, int col) {
  int ks = col >> 5, kk = col & 31;
  int h2 = (kk >> 3) & 1;
  int t2 = (kk & 7) | ((kk >> 4) << 3);
  return ((ks * 32 + h2 * 16 + m) << 4) + t2;
}

__device__ __forceinline__ v8f wmma32(v16h a, v16h b, v8f c) {
  return __builtin_amdgcn_wmma_f32_16x16x32_f16(false, a, false, b, (short)0, c,
                                                false, false);
}

__device__ __forceinline__ v8f zero8() {
  v8f z = {0.f, 0.f, 0.f, 0.f, 0.f, 0.f, 0.f, 0.f};
  return z;
}
__device__ __forceinline__ v8f splat8(float x) {
  v8f z = {x, x, x, x, x, x, x, x};
  return z;
}

__device__ __forceinline__ float siluf(float x) { return x / (1.0f + __expf(-x)); }
__device__ __forceinline__ float sigmf(float x) { return 1.0f / (1.0f + __expf(-x)); }

// A-fragment (16x32) from f32 row-major memory via 4x b128 loads.
// rowbase must be 16B aligned, k0 multiple of 32.
__device__ __forceinline__ v16h a_from_f32(const float* rowbase, int k0, int half) {
  float4 p0 = *(const float4*)(rowbase + k0 + half * 8);
  float4 p1 = *(const float4*)(rowbase + k0 + half * 8 + 4);
  float4 q0 = *(const float4*)(rowbase + k0 + 16 + half * 8);
  float4 q1 = *(const float4*)(rowbase + k0 + 16 + half * 8 + 4);
  v16h a;
  a[0] = (_Float16)p0.x;  a[1] = (_Float16)p0.y;
  a[2] = (_Float16)p0.z;  a[3] = (_Float16)p0.w;
  a[4] = (_Float16)p1.x;  a[5] = (_Float16)p1.y;
  a[6] = (_Float16)p1.z;  a[7] = (_Float16)p1.w;
  a[8]  = (_Float16)q0.x; a[9]  = (_Float16)q0.y;
  a[10] = (_Float16)q0.z; a[11] = (_Float16)q0.w;
  a[12] = (_Float16)q1.x; a[13] = (_Float16)q1.y;
  a[14] = (_Float16)q1.z; a[15] = (_Float16)q1.w;
  return a;
}

// Fragment-ordered LDS accessors: one 32B contiguous read per lane.
__device__ __forceinline__ v16h b_frag(const _Float16* wf, int NT, int kt, int nt,
                                       int lane) {
  return *(const v16h*)(wf + (((kt * NT + nt) * 32 + lane) << 4));
}
__device__ __forceinline__ v16h a_frag(const _Float16* st, int ks, int lane) {
  return *(const v16h*)(st + ((ks * 32 + lane) << 4));
}

// Cooperative fill of a fragment-ordered f16 weight image of W (K x Nn, f32).
__device__ __forceinline__ void fill_frag(const float* __restrict__ W,
                                          _Float16* __restrict__ dst, int K,
                                          int Nn) {
  int NT = Nn >> 4;
  int total = K * Nn;
  for (int i = threadIdx.x; i < total; i += blockDim.x) {
    int t = i & 15;
    int l = (i >> 4) & 31;
    int frag = i >> 9;  // kt*NT + nt
    int nt = frag % NT, kt = frag / NT;
    int half = l >> 4, n = l & 15;
    int k = kt * 32 + kmap(t, half);
    dst[i] = (_Float16)W[k * Nn + nt * 16 + n];
  }
}

// 64-wide two k-step GEMM micro-kernel: all 8 B fragments prefetched.
__device__ __forceinline__ void gemm64(v16h a0, v16h a1, const _Float16* wf,
                                       int lane, v8f acc[4]) {
  v16h bf[8];
#pragma unroll
  for (int ks = 0; ks < 2; ++ks)
#pragma unroll
    for (int nt = 0; nt < 4; ++nt) bf[ks * 4 + nt] = b_frag(wf, 4, ks, nt, lane);
#pragma unroll
  for (int nt = 0; nt < 4; ++nt) acc[nt] = wmma32(a0, bf[nt], acc[nt]);
#pragma unroll
  for (int nt = 0; nt < 4; ++nt) acc[nt] = wmma32(a1, bf[4 + nt], acc[nt]);
}

// ---------------------------------------------------------------- init
__global__ void k_init(const float* __restrict__ rb, const float* __restrict__ cutoff_w,
                       float* __restrict__ rbw, float* __restrict__ agg,
                       float* __restrict__ node_acc) {
  int tid = blockIdx.x * blockDim.x + threadIdx.x;
  int stride = gridDim.x * blockDim.x;
  for (int i = tid; i < N_EDGES * NORB; i += stride) rbw[i] = rb[i] * cutoff_w[i / NORB];
  for (int i = tid; i < N_EDGES * CDIM; i += stride) agg[i] = 0.f;
  for (int i = tid; i < N_NODES * CDIM; i += stride) node_acc[i] = 0.f;
}

// ------------------------------------------------- node MLP: x @ w_node + b
__global__ void __launch_bounds__(256)
k_node(const float* __restrict__ x, const float* __restrict__ w_node,
       const float* __restrict__ b_node, float* __restrict__ x1,
       float* __restrict__ xkg) {
  __shared__ __attribute__((aligned(32))) _Float16 wlf[HDIM * HDIM];
  fill_frag(w_node, wlf, HDIM, HDIM);
  __syncthreads();
  int wave = threadIdx.x >> 5, lane = threadIdx.x & 31;
  int half = lane >> 4, n = lane & 15, mloc = lane & 15;
  int tile = blockIdx.x * 8 + wave;
  if (tile >= N_NODES / 16) return;
  int row0 = tile * 16;

  v8f acc[8];
#pragma unroll
  for (int nt = 0; nt < 8; ++nt) acc[nt] = splat8(b_node[nt * 16 + n]);
  const float* rowbase = x + (size_t)(row0 + mloc) * HDIM;
#pragma unroll
  for (int ks = 0; ks < 4; ++ks) {
    v16h a = a_from_f32(rowbase, ks * 32, half);
    v16h bf[8];
#pragma unroll
    for (int nt = 0; nt < 8; ++nt) bf[nt] = b_frag(wlf, 8, ks, nt, lane);
#pragma unroll
    for (int nt = 0; nt < 8; ++nt) acc[nt] = wmma32(a, bf[nt], acc[nt]);
  }
#pragma unroll
  for (int nt = 0; nt < 8; ++nt)
#pragma unroll
    for (int v = 0; v < 8; ++v) {
      int col = nt * 16 + n;
      size_t r = (size_t)(row0 + 8 * half + v);
      float val = acc[nt][v];
      if (col < CDIM) x1[r * CDIM + col] = val;
      else            xkg[r * CDIM + (col - CDIM)] = sigmf(val);
    }
}

// ------------------------- edge orbital MLP: silu(silu(cji@w_c1)@w_c2) -> f16
__global__ void __launch_bounds__(256)
k_edge_mlp(const float* __restrict__ cji, const float* __restrict__ w_c1,
           const float* __restrict__ w_c2, _Float16* __restrict__ cji2) {
  __shared__ __attribute__((aligned(32))) _Float16 w1f[CDIM * CDIM];
  __shared__ __attribute__((aligned(32))) _Float16 w2f[CDIM * CDIM];
  __shared__ __attribute__((aligned(32))) _Float16 stage[8][16 * CDIM];
  fill_frag(w_c1, w1f, CDIM, CDIM);
  fill_frag(w_c2, w2f, CDIM, CDIM);
  __syncthreads();
  const int Mrows = N_EDGES * NORB;
  int wave = threadIdx.x >> 5, lane = threadIdx.x & 31;
  int half = lane >> 4, n = lane & 15, mloc = lane & 15;
  int tile = blockIdx.x * 8 + wave;
  if (tile * 16 >= Mrows) return;
  int row0 = tile * 16;
  int myrow = row0 + mloc;
  const float* rowbase =
      cji + (size_t)(myrow < Mrows ? myrow : (Mrows - 1)) * CDIM;
  _Float16* st = stage[wave];

  v8f acc[4];
#pragma unroll
  for (int nt = 0; nt < 4; ++nt) acc[nt] = zero8();
  {
    v16h a0 = a_from_f32(rowbase, 0, half);
    v16h a1 = a_from_f32(rowbase, 32, half);
    gemm64(a0, a1, w1f, lane, acc);
  }
  // silu -> fragment-ordered stage
#pragma unroll
  for (int nt = 0; nt < 4; ++nt)
#pragma unroll
    for (int v = 0; v < 8; ++v)
      st[frag_slot(8 * half + v, nt * 16 + n)] = (_Float16)siluf(acc[nt][v]);
  __builtin_amdgcn_wave_barrier();

  v8f acc2[4];
#pragma unroll
  for (int nt = 0; nt < 4; ++nt) acc2[nt] = zero8();
  {
    v16h a0 = a_frag(st, 0, lane);
    v16h a1 = a_frag(st, 1, lane);
    gemm64(a0, a1, w2f, lane, acc2);
  }
  __builtin_amdgcn_wave_barrier();
  // silu -> row-major stage, then coalesced v2h global stores
#pragma unroll
  for (int nt = 0; nt < 4; ++nt)
#pragma unroll
    for (int v = 0; v < 8; ++v)
      st[(8 * half + v) * CDIM + nt * 16 + n] = (_Float16)siluf(acc2[nt][v]);
  __builtin_amdgcn_wave_barrier();
#pragma unroll
  for (int m = 0; m < 16; ++m) {
    int r = row0 + m;
    if (r < Mrows)
      *(v2h*)(cji2 + (size_t)r * CDIM + 2 * lane) =
          *(const v2h*)(st + m * CDIM + 2 * lane);
  }
}

// -------------------- triplets: gather, weighted sum over d, normalize, scatter
__global__ void __launch_bounds__(256)
k_triplet(const _Float16* __restrict__ cji2, const float* __restrict__ rbw,
          const float* __restrict__ shb, const int* __restrict__ tri_k,
          const int* __restrict__ e_kj, const int* __restrict__ e_ji,
          const float* __restrict__ xkg, float* __restrict__ agg) {
  int t = blockIdx.x * 8 + (threadIdx.x >> 5);
  int lane = threadIdx.x & 31;
  if (t >= N_TRI) return;
  int ekj = e_kj[t], eji = e_ji[t], k = tri_k[t];
  // lane owns channels 2*lane and 2*lane+1 -> all gathers are dword loads
  const v2h* cb = (const v2h*)(cji2 + (size_t)ekj * NORB * CDIM);
  const float* rw = rbw + (size_t)ekj * NORB;
  const float* sh = shb + (size_t)t * NORB;
  float a0 = 0.f, a1 = 0.f;
#pragma unroll
  for (int d = 0; d < NORB; ++d) {
    float coef = rw[d] * sh[d];
    v2h p = cb[d * 32 + lane];
    a0 += coef * (float)p[0];
    a1 += coef * (float)p[1];
  }
  float ss = a0 * a0 + a1 * a1;
#pragma unroll
  for (int off = 16; off; off >>= 1) ss += __shfl_xor(ss, off, 32);
  float inv = 1.0f / fmaxf(sqrtf(ss), 1e-12f);
  float2 g = *(const float2*)(xkg + (size_t)k * CDIM + 2 * lane);
  atomicAdd(&agg[(size_t)eji * CDIM + 2 * lane],     a0 * inv * g.x);
  atomicAdd(&agg[(size_t)eji * CDIM + 2 * lane + 1], a1 * inv * g.y);
}

// --- fused per-edge: triplet MLP, cji update + lcao reduce, normalize, w_basis,
//     pair MLP, message scatter.  2 waves/block, 56KB static LDS.
__global__ void __launch_bounds__(64)
k_edge_fused(const float* __restrict__ agg, const float* __restrict__ w_t1,
             const float* __restrict__ b_t1, const float* __restrict__ w_t2,
             const float* __restrict__ b_t2, const float* __restrict__ w_basis,
             const float* __restrict__ w_n1, const float* __restrict__ b_n1,
             const float* __restrict__ w_n2, const float* __restrict__ b_n2,
             const _Float16* __restrict__ cji2, const float* __restrict__ rbw,
             const float* __restrict__ x1, const int* __restrict__ idx_i,
             const int* __restrict__ idx_j, float* __restrict__ node_acc) {
  __shared__ __attribute__((aligned(32))) _Float16 wt1f[CDIM * CDIM];
  __shared__ __attribute__((aligned(32))) _Float16 wt2f[CDIM * CDIM];
  __shared__ __attribute__((aligned(32))) _Float16 wbf[CDIM * CDIM];
  __shared__ __attribute__((aligned(32))) _Float16 wn2f[CDIM * CDIM];
  __shared__ __attribute__((aligned(32))) _Float16 wn1f[HDIM * CDIM];
  __shared__ __attribute__((aligned(32))) char stg[2][16 * CDIM * 4];
  fill_frag(w_t1, wt1f, CDIM, CDIM);
  fill_frag(w_t2, wt2f, CDIM, CDIM);
  fill_frag(w_basis, wbf, CDIM, CDIM);
  fill_frag(w_n2, wn2f, CDIM, CDIM);
  fill_frag(w_n1, wn1f, HDIM, CDIM);
  __syncthreads();
  int wave = threadIdx.x >> 5, lane = threadIdx.x & 31;
  int half = lane >> 4, n = lane & 15, mloc = lane & 15;
  int tile = blockIdx.x * 2 + wave;
  if (tile >= N_EDGES / 16) return;
  int e0 = tile * 16;
  _Float16* sth = (_Float16*)stg[wave];  // f16 fragment view (2KB used)
  float*    stf = (float*)stg[wave];     // f32 row-major view (4KB used)

  // ---- three_body_w = silu(silu(agg@w_t1+b_t1)@w_t2+b_t2)
  v8f acc[4];
#pragma unroll
  for (int nt = 0; nt < 4; ++nt) acc[nt] = splat8(b_t1[nt * 16 + n]);
  {
    const float* rowbase = agg + (size_t)(e0 + mloc) * CDIM;
    v16h a0 = a_from_f32(rowbase, 0, half);
    v16h a1 = a_from_f32(rowbase, 32, half);
    gemm64(a0, a1, wt1f, lane, acc);
  }
#pragma unroll
  for (int nt = 0; nt < 4; ++nt)
#pragma unroll
    for (int v = 0; v < 8; ++v)
      sth[frag_slot(8 * half + v, nt * 16 + n)] = (_Float16)siluf(acc[nt][v]);
  __builtin_amdgcn_wave_barrier();
  v8f tbw[4];
#pragma unroll
  for (int nt = 0; nt < 4; ++nt) tbw[nt] = splat8(b_t2[nt * 16 + n]);
  {
    v16h a0 = a_frag(sth, 0, lane);
    v16h a1 = a_frag(sth, 1, lane);
    gemm64(a0, a1, wt2f, lane, tbw);
  }
  __builtin_amdgcn_wave_barrier();
#pragma unroll
  for (int nt = 0; nt < 4; ++nt)
#pragma unroll
    for (int v = 0; v < 8; ++v) tbw[nt][v] = siluf(tbw[nt][v]);

  // ---- s[m,h] = sum_d rbw[e,d]*cji2[e,d,h]  (pair-channel dword gathers)
#pragma unroll
  for (int m = 0; m < 16; ++m) {
    int e = e0 + m;
    const v2h* cb = (const v2h*)(cji2 + (size_t)e * NORB * CDIM);
    const float* rw = rbw + (size_t)e * NORB;
    float s0 = 0.f, s1 = 0.f;
#pragma unroll
    for (int d = 0; d < NORB; ++d) {
      float c = rw[d];
      v2h p = cb[d * 32 + lane];
      s0 += c * (float)p[0];
      s1 += c * (float)p[1];
    }
    *(float2*)(stf + m * CDIM + 2 * lane) = make_float2(s0, s1);
  }
  __builtin_amdgcn_wave_barrier();

  // ---- lcao_pre = (1+tbw)*s; row-normalize; stage as f16 fragments
  float ln[4][8];
#pragma unroll
  for (int v = 0; v < 8; ++v) {
    float sv[4];
    float ss = 0.f;
#pragma unroll
    for (int nt = 0; nt < 4; ++nt) {
      float s = stf[(8 * half + v) * CDIM + nt * 16 + n];
      s *= (1.0f + tbw[nt][v]);
      sv[nt] = s;
      ss += s * s;
    }
#pragma unroll
    for (int off = 8; off; off >>= 1) ss += __shfl_xor(ss, off, 32);
    float inv = 1.0f / fmaxf(sqrtf(ss), 1e-12f);
#pragma unroll
    for (int nt = 0; nt < 4; ++nt) ln[nt][v] = sv[nt] * inv;
  }
  __builtin_amdgcn_wave_barrier();
#pragma unroll
  for (int nt = 0; nt < 4; ++nt)
#pragma unroll
    for (int v = 0; v < 8; ++v)
      sth[frag_slot(8 * half + v, nt * 16 + n)] = (_Float16)ln[nt][v];
  __builtin_amdgcn_wave_barrier();

  // ---- lcao_w = ln @ w_basis
  v8f lw[4];
#pragma unroll
  for (int nt = 0; nt < 4; ++nt) lw[nt] = zero8();
  {
    v16h a0 = a_frag(sth, 0, lane);
    v16h a1 = a_frag(sth, 1, lane);
    gemm64(a0, a1, wbf, lane, lw);
  }
  __builtin_amdgcn_wave_barrier();

  // ---- fnode = silu(silu([x1[i],x1[j]]@w_n1+b_n1)@w_n2+b_n2)
  int ei = idx_i[e0 + mloc];
  int ej = idx_j[e0 + mloc];
  v8f fn[4];
#pragma unroll
  for (int nt = 0; nt < 4; ++nt) fn[nt] = splat8(b_n1[nt * 16 + n]);
#pragma unroll
  for (int kp = 0; kp < 2; ++kp) {  // kp=0: x1[i] (K 0..63), kp=1: x1[j] (K 64..127)
    const float* rowb = (kp == 0) ? (x1 + (size_t)ei * CDIM)
                                  : (x1 + (size_t)ej * CDIM);
    v16h a0 = a_from_f32(rowb, 0, half);
    v16h a1 = a_from_f32(rowb, 32, half);
    v16h bf[8];
#pragma unroll
    for (int ks = 0; ks < 2; ++ks)
#pragma unroll
      for (int nt = 0; nt < 4; ++nt)
        bf[ks * 4 + nt] = b_frag(wn1f, 4, kp * 2 + ks, nt, lane);
#pragma unroll
    for (int nt = 0; nt < 4; ++nt) fn[nt] = wmma32(a0, bf[nt], fn[nt]);
#pragma unroll
    for (int nt = 0; nt < 4; ++nt) fn[nt] = wmma32(a1, bf[4 + nt], fn[nt]);
  }
#pragma unroll
  for (int nt = 0; nt < 4; ++nt)
#pragma unroll
    for (int v = 0; v < 8; ++v)
      sth[frag_slot(8 * half + v, nt * 16 + n)] = (_Float16)siluf(fn[nt][v]);
  __builtin_amdgcn_wave_barrier();
  v8f fn2[4];
#pragma unroll
  for (int nt = 0; nt < 4; ++nt) fn2[nt] = splat8(b_n2[nt * 16 + n]);
  {
    v16h a0 = a_frag(sth, 0, lane);
    v16h a1 = a_frag(sth, 1, lane);
    gemm64(a0, a1, wn2f, lane, fn2);
  }

  // ---- msg = lcao_w * silu(fn2); scatter-add into node_acc[idx_i]
#pragma unroll
  for (int v = 0; v < 8; ++v) {
    int e = e0 + 8 * half + v;
    int dst = idx_i[e];
#pragma unroll
    for (int nt = 0; nt < 4; ++nt) {
      float msg = lw[nt][v] * siluf(fn2[nt][v]);
      atomicAdd(&node_acc[(size_t)dst * CDIM + nt * 16 + n], msg);
    }
  }
}

// ------------------------------------ out = x + node_acc @ w_out
__global__ void __launch_bounds__(256)
k_out(const float* __restrict__ x, const float* __restrict__ node_acc,
      const float* __restrict__ w_out, float* __restrict__ out) {
  __shared__ __attribute__((aligned(32))) _Float16 wlf[CDIM * HDIM];
  fill_frag(w_out, wlf, CDIM, HDIM);
  __syncthreads();
  int wave = threadIdx.x >> 5, lane = threadIdx.x & 31;
  int half = lane >> 4, n = lane & 15, mloc = lane & 15;
  int tile = blockIdx.x * 8 + wave;
  if (tile >= N_NODES / 16) return;
  int row0 = tile * 16;
  v8f acc[8];
#pragma unroll
  for (int nt = 0; nt < 8; ++nt) acc[nt] = zero8();
  const float* rowbase = node_acc + (size_t)(row0 + mloc) * CDIM;
#pragma unroll
  for (int ks = 0; ks < 2; ++ks) {
    v16h a = a_from_f32(rowbase, ks * 32, half);
    v16h bf[8];
#pragma unroll
    for (int nt = 0; nt < 8; ++nt) bf[nt] = b_frag(wlf, 8, ks, nt, lane);
#pragma unroll
    for (int nt = 0; nt < 8; ++nt) acc[nt] = wmma32(a, bf[nt], acc[nt]);
  }
#pragma unroll
  for (int nt = 0; nt < 8; ++nt)
#pragma unroll
    for (int v = 0; v < 8; ++v) {
      size_t r = (size_t)(row0 + 8 * half + v);
      int col = nt * 16 + n;
      out[r * HDIM + col] = x[r * HDIM + col] + acc[nt][v];
    }
}

extern "C" void kernel_launch(void* const* d_in, const int* in_sizes, int n_in,
                              void* d_out, int out_size, void* d_ws, size_t ws_size,
                              hipStream_t stream) {
  (void)in_sizes; (void)n_in; (void)out_size; (void)ws_size;
  const float* x        = (const float*)d_in[0];
  const float* cji      = (const float*)d_in[1];
  const float* cutoff_w = (const float*)d_in[2];
  const float* rb       = (const float*)d_in[3];
  const float* shb      = (const float*)d_in[4];
  const int*   idx_i    = (const int*)d_in[5];
  const int*   idx_j    = (const int*)d_in[6];
  const int*   tri_k    = (const int*)d_in[7];
  const int*   e_kj     = (const int*)d_in[8];
  const int*   e_ji     = (const int*)d_in[9];
  const float* w_node   = (const float*)d_in[10];
  const float* b_node   = (const float*)d_in[11];
  const float* w_c1     = (const float*)d_in[12];
  const float* w_c2     = (const float*)d_in[13];
  const float* w_t1     = (const float*)d_in[14];
  const float* b_t1     = (const float*)d_in[15];
  const float* w_t2     = (const float*)d_in[16];
  const float* b_t2     = (const float*)d_in[17];
  const float* w_basis  = (const float*)d_in[18];
  const float* w_n1     = (const float*)d_in[19];
  const float* b_n1     = (const float*)d_in[20];
  const float* w_n2     = (const float*)d_in[21];
  const float* b_n2     = (const float*)d_in[22];
  const float* w_out    = (const float*)d_in[23];
  float* out = (float*)d_out;

  char* ws = (char*)d_ws;
  size_t off = 0;
  auto alloc = [&](size_t bytes) {
    void* p = ws + off;
    off = (off + bytes + 255) & ~(size_t)255;
    return p;
  };
  float*    x1       = (float*)alloc((size_t)N_NODES * CDIM * 4);
  float*    xkg      = (float*)alloc((size_t)N_NODES * CDIM * 4);
  float*    rbw      = (float*)alloc((size_t)N_EDGES * NORB * 4);
  float*    aggp     = (float*)alloc((size_t)N_EDGES * CDIM * 4);
  float*    node_acc = (float*)alloc((size_t)N_NODES * CDIM * 4);
  _Float16* cji2     = (_Float16*)alloc((size_t)N_EDGES * NORB * CDIM * 2);

  k_init<<<2048, 256, 0, stream>>>(rb, cutoff_w, rbw, aggp, node_acc);

  int node_tiles = N_NODES / 16;  // 625
  k_node<<<(node_tiles + 7) / 8, 256, 0, stream>>>(x, w_node, b_node, x1, xkg);

  int orb_tiles = (N_EDGES * NORB + 15) / 16;  // 56250
  k_edge_mlp<<<(orb_tiles + 7) / 8, 256, 0, stream>>>(cji, w_c1, w_c2, cji2);

  k_triplet<<<N_TRI / 8, 256, 0, stream>>>(cji2, rbw, shb, tri_k, e_kj, e_ji,
                                           xkg, aggp);

  int edge_tiles = N_EDGES / 16;  // 6250
  k_edge_fused<<<(edge_tiles + 1) / 2, 64, 0, stream>>>(
      aggp, w_t1, b_t1, w_t2, b_t2, w_basis, w_n1, b_n1, w_n2, b_n2, cji2, rbw,
      x1, idx_i, idx_j, node_acc);

  k_out<<<(node_tiles + 7) / 8, 256, 0, stream>>>(x, node_acc, w_out, out);
}
